// MultiHeadAttention_45088566673628
// MI455X (gfx1250) — matrix operand order repaired and checked by reading, not compiled, not run on hardware
//
#include <hip/hip_runtime.h>
#include <hip/hip_bf16.h>

#define B_  4
#define T_  2048
#define D_  1024
#define H_  16
#define DK_ 64
#define M_  (B_ * T_)

typedef unsigned short u16;
typedef __bf16 bf16x16 __attribute__((ext_vector_type(16)));
typedef float  f32x8   __attribute__((ext_vector_type(8)));

union Frag16 {
    bf16x16 v;
    uint4   q[2];
    u16     s[16];
};

static __device__ __forceinline__ u16 f32_to_bf16(float f) {
    union { float f; unsigned u; } a; a.f = f;
    unsigned r = a.u + 0x7FFFu + ((a.u >> 16) & 1u);   // round-to-nearest-even
    return (u16)(r >> 16);
}

static __device__ __forceinline__ f32x8 wmma_bf16(const Frag16& a, const Frag16& b, f32x8 c) {
    // D = A(16x32 bf16) * B(32x16 bf16) + C(16x16 f32)
    return __builtin_amdgcn_wmma_f32_16x16x32_bf16(false, a.v, false, b.v, (short)0, c, false, false);
}

// ---------------------------------------------------------------------------
// f32 -> bf16 bulk convert (4 elements / thread)
// ---------------------------------------------------------------------------
__global__ __launch_bounds__(256) void cvt_f32_bf16(const float* __restrict__ in,
                                                    u16* __restrict__ out, int n4) {
    int i = blockIdx.x * blockDim.x + threadIdx.x;
    if (i < n4) {
        float4 v = ((const float4*)in)[i];
        union { u16 s[4]; uint2 u; } t;
        t.s[0] = f32_to_bf16(v.x); t.s[1] = f32_to_bf16(v.y);
        t.s[2] = f32_to_bf16(v.z); t.s[3] = f32_to_bf16(v.w);
        ((uint2*)out)[i] = t.u;
    }
}

// ---------------------------------------------------------------------------
// Tiled bf16 GEMM: C[M,N] = X[M,K] * W[K,N] + bias[N]
// 256 threads = 8 waves; block tile 128x128; K-step 32.
// Wave layout: waves (wave>>1) -> 4 M slots of 32 rows; (wave&1) -> 2 N slots of 64 cols.
// mode 0: out bf16 scatter [B,H,T,DK]  (Q)
// mode 1: out bf16 scatter [B,H,T,DK]  (K)
// mode 2: out bf16 scatter [B,H,DK,T]  (V transposed)
// mode 3: out f32 row-major [M,N]      (final projection)
// ---------------------------------------------------------------------------
__global__ __launch_bounds__(256) void gemm_bf16(const u16* __restrict__ X,
                                                 const u16* __restrict__ W,
                                                 const float* __restrict__ bias,
                                                 void* __restrict__ outp,
                                                 int mode) {
    __shared__ __align__(16) u16 xs[128 * 40];   // [row][k]  k-stride padded to 40
    __shared__ __align__(16) u16 wsm[128 * 40];  // [n][k]    transposed W tile

    const int tid  = threadIdx.x;
    const int wave = tid >> 5, lane = tid & 31, hi = lane >> 4, lo = lane & 15;
    const int mblk = blockIdx.y * 128, nblk = blockIdx.x * 128;
    const int wm = (wave >> 1) * 32;   // wave M offset inside tile
    const int wn = (wave & 1) * 64;    // wave N offset inside tile

    const f32x8 z = {0.f, 0.f, 0.f, 0.f, 0.f, 0.f, 0.f, 0.f};
    f32x8 acc[2][4];
    #pragma unroll
    for (int i = 0; i < 2; ++i)
        #pragma unroll
        for (int j = 0; j < 4; ++j) acc[i][j] = z;

    const int xrow = tid >> 1, xkh = (tid & 1) * 16;   // X stage: 128 rows x 32 k
    const int wkk = tid >> 3, wnb = (tid & 7) * 16;    // W stage: 32 k x 128 n

    for (int k0 = 0; k0 < D_; k0 += 32) {
        __syncthreads();
        { // stage X tile (row-major, bf16)
            const uint4* src = (const uint4*)(X + (size_t)(mblk + xrow) * D_ + k0 + xkh);
            uint4 v0 = src[0], v1 = src[1];
            *(uint4*)(xs + xrow * 40 + xkh)     = v0;
            *(uint4*)(xs + xrow * 40 + xkh + 8) = v1;
        }
        { // stage W tile transposed: wsm[n][k]
            union { uint4 q[2]; u16 s[16]; } t;
            const uint4* src = (const uint4*)(W + (size_t)(k0 + wkk) * D_ + nblk + wnb);
            t.q[0] = src[0]; t.q[1] = src[1];
            #pragma unroll
            for (int e = 0; e < 16; ++e) wsm[(wnb + e) * 40 + wkk] = t.s[e];
        }
        if (k0 + 32 < D_) {
            __builtin_prefetch(X + (size_t)(mblk + xrow) * D_ + k0 + 32 + xkh, 0, 1);
            __builtin_prefetch(W + (size_t)(k0 + 32 + wkk) * D_ + nblk + wnb, 0, 1);
        }
        __syncthreads();

        Frag16 a[2];
        #pragma unroll
        for (int mt = 0; mt < 2; ++mt) {
            // A frag: lane<16 -> row=lane, K {0..7,16..23}; lane>=16 -> K {8..15,24..31}
            const u16* rb = xs + (wm + mt * 16 + lo) * 40;
            a[mt].q[0] = *(const uint4*)(rb + hi * 8);
            a[mt].q[1] = *(const uint4*)(rb + 16 + hi * 8);
        }
        #pragma unroll
        for (int nt = 0; nt < 4; ++nt) {
            // B frag: lane<16 -> col=lane, K {0..7,16..23}; lane>=16 -> col=lane-16, K {8..15,24..31}
            Frag16 b;
            const u16* cb = wsm + (wn + nt * 16 + lo) * 40 + hi * 8;
            b.q[0] = *(const uint4*)(cb);
            b.q[1] = *(const uint4*)(cb + 16);
            #pragma unroll
            for (int mt = 0; mt < 2; ++mt)
                acc[mt][nt] = wmma_bf16(a[mt], b, acc[mt][nt]);
        }
    }

    // epilogue: C frag element r is row r + 8*hi, col = lo
    #pragma unroll
    for (int nt = 0; nt < 4; ++nt) {
        const int n  = nblk + wn + nt * 16 + lo;
        const float bv = bias[n];
        #pragma unroll
        for (int mt = 0; mt < 2; ++mt) {
            #pragma unroll
            for (int r = 0; r < 8; ++r) {
                const int m = mblk + wm + mt * 16 + r + 8 * hi;
                const float val = acc[mt][nt][r] + bv;
                if (mode == 3) {
                    ((float*)outp)[(size_t)m * D_ + n] = val;
                } else {
                    const int b_ = m >> 11, t = m & (T_ - 1);   // T_ = 2048
                    const int h = n >> 6, dk = n & 63;          // DK_ = 64
                    size_t idx;
                    if (mode == 2) idx = (((size_t)(b_ * H_ + h)) * DK_ + dk) * T_ + t;
                    else           idx = (((size_t)(b_ * H_ + h)) * T_ + t) * DK_ + dk;
                    ((u16*)outp)[idx] = f32_to_bf16(val);
                }
            }
        }
    }
}

// ---------------------------------------------------------------------------
// Flash attention (causal), bf16 WMMA, f32 online softmax.
// Grid: (T/128, B*H), 256 threads = 8 waves; each wave owns 16 query rows.
// Q,K: [B,H,T,DK] bf16; Vt: [B,H,DK,T] bf16; AO: [B,T,D] bf16.
// ---------------------------------------------------------------------------
__global__ __launch_bounds__(256) void flash_attn(const u16* __restrict__ Q,
                                                  const u16* __restrict__ K,
                                                  const u16* __restrict__ Vt,
                                                  u16* __restrict__ AO) {
    __shared__ __align__(16) u16 Pbuf[8 * 16 * 40];   // per-wave 16x32 P tile (padded)

    const int tid  = threadIdx.x;
    const int wave = tid >> 5, lane = tid & 31, hi = lane >> 4, lo = lane & 15;
    const int bh = blockIdx.y;
    const int b_ = bh >> 4, h = bh & 15;
    const int q0 = blockIdx.x * 128 + wave * 16;

    const u16* Qh = Q  + (size_t)bh * T_ * DK_;
    const u16* Kh = K  + (size_t)bh * T_ * DK_;
    const u16* Vh = Vt + (size_t)bh * DK_ * T_;
    u16* Pw = Pbuf + wave * (16 * 40);

    // Q A-fragments for both 32-wide dk chunks
    Frag16 aq[2];
    const u16* qr = Qh + (size_t)(q0 + lo) * DK_;
    #pragma unroll
    for (int c = 0; c < 2; ++c) {
        aq[c].q[0] = *(const uint4*)(qr + c * 32 + hi * 8);
        aq[c].q[1] = *(const uint4*)(qr + c * 32 + 16 + hi * 8);
    }

    const f32x8 z = {0.f, 0.f, 0.f, 0.f, 0.f, 0.f, 0.f, 0.f};
    f32x8 o[4] = {z, z, z, z};
    float m[8], l[8];
    const float ninf = -__builtin_inff();
    #pragma unroll
    for (int r = 0; r < 8; ++r) { m[r] = ninf; l[r] = 0.f; }

    for (int j0 = 0; j0 <= q0 + 15; j0 += 32) {
        // ---- scores: S = Q * K^T over 32 keys (two 16-key C tiles) ----
        f32x8 s0 = z, s1 = z;
        #pragma unroll
        for (int c = 0; c < 2; ++c) {
            Frag16 bk0, bk1;
            const u16* kr0 = Kh + (size_t)(j0 + lo) * DK_ + c * 32 + hi * 8;
            const u16* kr1 = Kh + (size_t)(j0 + 16 + lo) * DK_ + c * 32 + hi * 8;
            bk0.q[0] = *(const uint4*)(kr0); bk0.q[1] = *(const uint4*)(kr0 + 16);
            bk1.q[0] = *(const uint4*)(kr1); bk1.q[1] = *(const uint4*)(kr1 + 16);
            s0 = wmma_bf16(aq[c], bk0, s0);
            s1 = wmma_bf16(aq[c], bk1, s1);
        }
        __builtin_prefetch(Kh + (size_t)(j0 + 32 + lo) * DK_, 0, 1);

        // ---- online softmax update (per C-frag row r + 8*hi) ----
        #pragma unroll
        for (int r = 0; r < 8; ++r) {
            const int row = q0 + r + 8 * hi;
            float v0 = (j0 + lo      <= row) ? s0[r] * 0.125f : ninf;  // 1/sqrt(64)
            float v1 = (j0 + 16 + lo <= row) ? s1[r] * 0.125f : ninf;
            float mt = fmaxf(v0, v1);
            #pragma unroll
            for (int d = 1; d < 16; d <<= 1) mt = fmaxf(mt, __shfl_xor(mt, d));
            const float mn    = fmaxf(m[r], mt);
            const float alpha = __expf(m[r] - mn);
            const float p0 = __expf(v0 - mn);
            const float p1 = __expf(v1 - mn);
            float rs = p0 + p1;
            #pragma unroll
            for (int d = 1; d < 16; d <<= 1) rs += __shfl_xor(rs, d);
            l[r] = l[r] * alpha + rs;
            m[r] = mn;
            #pragma unroll
            for (int f = 0; f < 4; ++f) o[f][r] *= alpha;
            // C-layout -> LDS (row, col) for A-fragment re-load
            Pw[(r + 8 * hi) * 40 + lo]      = f32_to_bf16(p0);
            Pw[(r + 8 * hi) * 40 + 16 + lo] = f32_to_bf16(p1);
        }
        asm volatile("s_wait_dscnt 0" ::: "memory");  // cross-lane LDS RAW inside wave

        // ---- P as A-fragment ----
        Frag16 ap;
        const u16* prp = Pw + lo * 40;
        ap.q[0] = *(const uint4*)(prp + hi * 8);
        ap.q[1] = *(const uint4*)(prp + 16 + hi * 8);

        // ---- O += P * V  (V transposed: B frag contiguous along T) ----
        #pragma unroll
        for (int f = 0; f < 4; ++f) {
            Frag16 bv;
            const u16* vr = Vh + (size_t)(f * 16 + lo) * T_ + j0 + hi * 8;
            bv.q[0] = *(const uint4*)(vr);
            bv.q[1] = *(const uint4*)(vr + 16);
            o[f] = wmma_bf16(ap, bv, o[f]);
        }
    }

    // ---- normalize and write attention output [B,T,D] bf16 ----
    #pragma unroll
    for (int f = 0; f < 4; ++f) {
        #pragma unroll
        for (int r = 0; r < 8; ++r) {
            const int t = q0 + r + 8 * hi;
            AO[((size_t)(b_ * T_ + t)) * D_ + h * DK_ + f * 16 + lo] =
                f32_to_bf16(o[f][r] / l[r]);
        }
    }
}

// ---------------------------------------------------------------------------
extern "C" void kernel_launch(void* const* d_in, const int* in_sizes, int n_in,
                              void* d_out, int out_size, void* d_ws, size_t ws_size,
                              hipStream_t stream) {
    const float* x  = (const float*)d_in[0];
    // d_in[1] = causal mask (bool) -- computed analytically in-kernel
    const float* wq = (const float*)d_in[2];
    const float* bq = (const float*)d_in[3];
    const float* wk = (const float*)d_in[4];
    const float* bk = (const float*)d_in[5];
    const float* wv = (const float*)d_in[6];
    const float* bv = (const float*)d_in[7];
    const float* wo = (const float*)d_in[8];
    const float* bo = (const float*)d_in[9];

    char* ws = (char*)d_ws;
    const size_t MB = 1ull << 20;
    u16* xb  = (u16*)(ws + 0);        // 16 MB  x in bf16 (reused as AO later)
    u16* wqb = (u16*)(ws + 16 * MB);  //  2 MB
    u16* wkb = (u16*)(ws + 18 * MB);
    u16* wvb = (u16*)(ws + 20 * MB);
    u16* wob = (u16*)(ws + 22 * MB);
    u16* Qb  = (u16*)(ws + 24 * MB);  // 16 MB [B,H,T,DK]
    u16* Kb  = (u16*)(ws + 40 * MB);  // 16 MB [B,H,T,DK]
    u16* Vtb = (u16*)(ws + 56 * MB);  // 16 MB [B,H,DK,T]
    u16* AOb = xb;                    // reuse x region after QKV projections

    const int nx = B_ * T_ * D_;   // 8388608
    const int nw = D_ * D_;        // 1048576
    cvt_f32_bf16<<<nx / 4 / 256, 256, 0, stream>>>(x,  xb,  nx / 4);
    cvt_f32_bf16<<<nw / 4 / 256, 256, 0, stream>>>(wq, wqb, nw / 4);
    cvt_f32_bf16<<<nw / 4 / 256, 256, 0, stream>>>(wk, wkb, nw / 4);
    cvt_f32_bf16<<<nw / 4 / 256, 256, 0, stream>>>(wv, wvb, nw / 4);
    cvt_f32_bf16<<<nw / 4 / 256, 256, 0, stream>>>(wo, wob, nw / 4);

    dim3 gg(D_ / 128, M_ / 128);   // (8, 64)
    gemm_bf16<<<gg, 256, 0, stream>>>(xb, wqb, bq, Qb, 0);
    gemm_bf16<<<gg, 256, 0, stream>>>(xb, wkb, bk, Kb, 1);
    gemm_bf16<<<gg, 256, 0, stream>>>(xb, wvb, bv, Vtb, 2);

    dim3 ga(T_ / 128, B_ * H_);    // (16, 64)
    flash_attn<<<ga, 256, 0, stream>>>(Qb, Kb, Vtb, AOb);

    gemm_bf16<<<gg, 256, 0, stream>>>(AOb, wob, bo, d_out, 3);

    (void)in_sizes; (void)n_in; (void)out_size; (void)ws_size;
}
